// PVConvBlock_75531294867876
// MI455X (gfx1250) — matrix operand order repaired and checked by reading, not compiled, not run on hardware
//
#include <hip/hip_runtime.h>
#include <hip/hip_bf16.h>
#include <cstdint>
#include <cstddef>

#define DXv 128
#define DYv 128
#define DZv 32
#define NVOX (DXv * DYv * DZv)   // 524288
#define NPTS 200000
#define CIN_ 32
#define COUT_ 64
#define BN_EPS_ 0.001f

typedef __bf16  v16bf __attribute__((ext_vector_type(16)));
typedef float   v8f   __attribute__((ext_vector_type(8)));
typedef unsigned int u32x4 __attribute__((ext_vector_type(4)));
typedef int     i32x8 __attribute__((ext_vector_type(8)));
typedef int     i32x4 __attribute__((ext_vector_type(4)));

union ABfrag { v16bf v; u32x4 u[2]; };

__device__ __forceinline__ unsigned short f2bf(float f) {
    return __builtin_bit_cast(unsigned short, (__bf16)f);
}

// ---- TDM availability (6-arg amdgpu-toolchain form) ----------------------
#if defined(__HIP_DEVICE_COMPILE__) && defined(__has_builtin)
#if __has_builtin(__builtin_amdgcn_tensor_load_to_lds) && __has_builtin(__builtin_amdgcn_s_wait_tensorcnt)
#define HAVE_TDM 1
#endif
#endif
#ifndef HAVE_TDM
#define HAVE_TDM 0
#endif

// ==========================================================================
// K1: point scatter  grid[v, c] += inputs[i, c]   (one thread per (i,c))
// ==========================================================================
__global__ void scatter_kernel(const float* __restrict__ X,
                               const int* __restrict__ vidx,
                               float* __restrict__ gridF) {
    int t = blockIdx.x * blockDim.x + threadIdx.x;
    if (t >= NPTS * CIN_) return;
    int i = t >> 5, c = t & 31;
    int x = vidx[i * 3 + 0], y = vidx[i * 3 + 1], z = vidx[i * 3 + 2];
    size_t v = (size_t)((x << 12) | (y << 5) | z);
    atomicAdd(gridF + v * CIN_ + c, X[t]);
}

// ==========================================================================
// K2: weight repack to bf16, column-major per tap: Wt[o][n][c]
// ==========================================================================
__global__ void prep_weights(const float* __restrict__ k1,
                             const float* __restrict__ k2,
                             const float* __restrict__ mw,
                             unsigned short* __restrict__ wt1,
                             unsigned short* __restrict__ wt2,
                             unsigned short* __restrict__ wtm) {
    const int N1 = 27 * 64 * 32, N2 = 27 * 64 * 64, NM = 64 * 32;
    int t = blockIdx.x * blockDim.x + threadIdx.x;
    if (t < N1) {
        int o = t / 2048, r = t % 2048, n = r / 32, c = r % 32;
        wt1[t] = f2bf(k1[(o * 32 + c) * 64 + n]);
    } else if (t < N1 + N2) {
        int u = t - N1;
        int o = u / 4096, r = u % 4096, n = r / 64, c = r % 64;
        wt2[u] = f2bf(k2[(o * 64 + c) * 64 + n]);
    } else if (t < N1 + N2 + NM) {
        int u = t - N1 - N2, n = u / 32, c = u % 32;
        wtm[u] = f2bf(mw[c * 64 + n]);
    }
}

// ==========================================================================
// K3: per voxel: pack 27-bit neighborhood occupancy mask + grid f32 -> bf16
// ==========================================================================
__global__ void prep_voxels(const float* __restrict__ gridF,
                            const float* __restrict__ ppv,
                            unsigned* __restrict__ gridB,
                            unsigned* __restrict__ maskb) {
    int v = blockIdx.x * blockDim.x + threadIdx.x;
    if (v >= NVOX) return;
    int z = v & 31, y = (v >> 5) & 127, x = v >> 12;
    unsigned m = 0;
    int o = 0;
    for (int ox = -1; ox <= 1; ++ox)
        for (int oy = -1; oy <= 1; ++oy)
            for (int oz = -1; oz <= 1; ++oz, ++o) {
                int nx = x + ox, ny = y + oy, nz = z + oz;
                if (nx >= 0 && nx < DXv && ny >= 0 && ny < DYv && nz >= 0 && nz < DZv) {
                    int nv = (nx << 12) | (ny << 5) | nz;
                    if (ppv[nv] != 0.0f) m |= (1u << o);
                }
            }
    maskb[v] = m;
    const float* src = gridF + (size_t)v * CIN_;
    unsigned* dst = gridB + (size_t)v * (CIN_ / 2);
#pragma unroll
    for (int c = 0; c < CIN_ / 2; ++c)
        dst[c] = (unsigned)f2bf(src[2 * c]) | ((unsigned)f2bf(src[2 * c + 1]) << 16);
}

// ==========================================================================
// K4: MLP via WMMA bf16 16x16x32, bias in C-init, relu+BN epilogue
// ==========================================================================
__global__ __launch_bounds__(256) void mlp_wmma(
        const float* __restrict__ X, const unsigned short* __restrict__ Wt,
        const float* __restrict__ bias, const float* __restrict__ gam,
        const float* __restrict__ bet, const float* __restrict__ mu,
        const float* __restrict__ var, float* __restrict__ out) {
    const int lane = threadIdx.x & 31, lc = lane & 15, hf = lane >> 4;
    const int gw = blockIdx.x * (blockDim.x >> 5) + (threadIdx.x >> 5);
    const int nwaves = gridDim.x * (blockDim.x >> 5);
    const int T = NPTS / 16;  // 12500
    for (int t = gw; t < T; t += nwaves) {
        const float* xp = X + (size_t)(t * 16 + lc) * CIN_;
        v16bf a;
#pragma unroll
        for (int e = 0; e < 8; ++e) {
            a[e]     = (__bf16)xp[hf * 8 + e];
            a[e + 8] = (__bf16)xp[16 + hf * 8 + e];
        }
#pragma unroll
        for (int nt = 0; nt < 4; ++nt) {
            int col = nt * 16 + lc;
            ABfrag b;
            const u32x4* wp = (const u32x4*)(Wt + col * 32 + hf * 8);
            b.u[0] = wp[0]; b.u[1] = wp[2];
            float bb = bias[col];
            v8f c;
#pragma unroll
            for (int r = 0; r < 8; ++r) c[r] = bb;
            c = __builtin_amdgcn_wmma_f32_16x16x32_bf16(false, a, false, b.v,
                                                        (short)0, c, false, false);
            float sc = rsqrtf(var[col] + BN_EPS_) * gam[col];
            float mm = mu[col], be = bet[col];
#pragma unroll
            for (int r = 0; r < 8; ++r) {
                float y = c[r];
                y = y > 0.0f ? y : 0.0f;
                y = (y - mm) * sc + be;
                out[(size_t)(t * 16 + r + hf * 8) * COUT_ + col] = y;
            }
        }
    }
}

// ==========================================================================
// K5/K6: masked 27-tap conv as WMMA GEMMs; weights resident in LDS via TDM
// ==========================================================================
template <int CIN, bool OUTF32>
__global__ __launch_bounds__(256) void conv_wmma(
        const unsigned* __restrict__ featB,     // bf16 rows, CIN/2 dwords
        const unsigned* __restrict__ maskb,
        const unsigned short* __restrict__ wt,  // bf16 [27][64][CIN]
        void* __restrict__ outp) {
    extern __shared__ char smem[];
    constexpr int WBYTES = 27 * 64 * CIN * 2;

#if HAVE_TDM
    if ((threadIdx.x >> 5) == 0) {
        unsigned long long ga = (unsigned long long)(uintptr_t)wt;
        const int ne = WBYTES / 8;  // 8-byte elements (data_size=3)
        u32x4 g0;
        g0[0] = 1u;                                   // count=1, user descriptor
        g0[1] = 0u;                                   // lds_addr = 0 (dyn-LDS base)
        g0[2] = (unsigned)(ga & 0xFFFFFFFFu);         // global_addr[31:0]
        g0[3] = (unsigned)((ga >> 32) & 0x01FFFFFFu)  // global_addr[56:32]
                | 0x80000000u;                        // type=2 ("image")
        i32x8 g1;
        g1[0] = 0x00030000;                            // data_size=8B
        g1[1] = (int)((unsigned)(ne & 0xFFFF) << 16);  // tensor_dim0 lo
        g1[2] = (int)(((unsigned)ne >> 16) | (1u << 16));  // dim0 hi | tensor_dim1=1
        g1[3] = (int)((unsigned)(ne & 0xFFFF) << 16);  // tile_dim0
        g1[4] = 1;                                     // tile_dim1=1
        g1[5] = ne;                                    // dim0 stride lo
        g1[6] = (int)(((unsigned)ne >> 16) | ((unsigned)(ne & 0xFFFF) << 16));
        g1[7] = (int)((unsigned)ne >> 16);
        i32x4 gz = (i32x4){0, 0, 0, 0};
        i32x8 gz8 = (i32x8){0, 0, 0, 0, 0, 0, 0, 0};
        __builtin_amdgcn_tensor_load_to_lds(g0, g1, gz, gz, gz8, 0);
        __builtin_amdgcn_s_wait_tensorcnt((short)0);
    }
#else
    {
        const u32x4* g = (const u32x4*)wt;
        u32x4* s4 = (u32x4*)smem;
        for (int i = threadIdx.x; i < WBYTES / 16; i += blockDim.x) s4[i] = g[i];
    }
#endif
    __syncthreads();

    const int lane = threadIdx.x & 31, lc = lane & 15, hf = lane >> 4;
    const int gw = blockIdx.x * (blockDim.x >> 5) + (threadIdx.x >> 5);
    const int nwaves = gridDim.x * (blockDim.x >> 5);
    const int T = NVOX / 16;
    const unsigned short* wl = (const unsigned short*)smem;

    for (int t = gw; t < T; t += nwaves) {
        int vrow = t * 16 + lc;
        const unsigned* fp = featB + (size_t)vrow * (CIN / 2);
        ABfrag afr[CIN / 32];
#pragma unroll
        for (int s = 0; s < CIN / 32; ++s) {
            afr[s].u[0] = *(const u32x4*)(fp + s * 16 + hf * 4);
            afr[s].u[1] = *(const u32x4*)(fp + s * 16 + 8 + hf * 4);
        }
        unsigned mb = maskb[vrow];
        v8f acc[4];
        const v8f zz = {0, 0, 0, 0, 0, 0, 0, 0};
#pragma unroll
        for (int nt = 0; nt < 4; ++nt) acc[nt] = zz;

        for (int o = 0; o < 27; ++o) {
            unsigned sel = ((mb >> o) & 1u) ? 0xFFFFFFFFu : 0u;
            ABfrag am[CIN / 32];
#pragma unroll
            for (int s = 0; s < CIN / 32; ++s) {
                am[s].u[0] = afr[s].u[0] & sel;
                am[s].u[1] = afr[s].u[1] & sel;
            }
#pragma unroll
            for (int nt = 0; nt < 4; ++nt) {
                int col = nt * 16 + lc;
                const unsigned short* bp = wl + (o * 64 + col) * CIN;
#pragma unroll
                for (int s = 0; s < CIN / 32; ++s) {
                    ABfrag b;
                    const u32x4* lp = (const u32x4*)(bp + s * 32 + hf * 8);
                    b.u[0] = lp[0]; b.u[1] = lp[2];
                    acc[nt] = __builtin_amdgcn_wmma_f32_16x16x32_bf16(
                        false, am[s].v, false, b.v, (short)0, acc[nt], false, false);
                }
            }
        }
#pragma unroll
        for (int nt = 0; nt < 4; ++nt)
#pragma unroll
            for (int r = 0; r < 8; ++r) {
                float y = acc[nt][r];
                y = y > 0.0f ? y : 0.0f;
                size_t off = (size_t)(t * 16 + r + hf * 8) * COUT_ + nt * 16 + lc;
                if (OUTF32) ((float*)outp)[off] = y;
                else        ((unsigned short*)outp)[off] = f2bf(y);
            }
    }
}

// ==========================================================================
// K7: trilinear sample of h2 and accumulate into d_out (thread = (pt, 16ch))
// ==========================================================================
__global__ void trilinear_add(const float* __restrict__ h2,
                              const float* __restrict__ pts,
                              float* __restrict__ out) {
    int tid = blockIdx.x * blockDim.x + threadIdx.x;
    int i = tid >> 2, g = tid & 3;
    if (i >= NPTS) return;
    float px = pts[i * 3 + 0], py = pts[i * 3 + 1], pz = pts[i * 3 + 2];
    int ix = min(max((int)floorf(px), 0), DXv - 2);
    int iy = min(max((int)floorf(py), 0), DYv - 2);
    int iz = min(max((int)floorf(pz), 0), DZv - 2);
    float fx = px - (float)ix, fy = py - (float)iy, fz = pz - (float)iz;
    float acc[16];
#pragma unroll
    for (int c = 0; c < 16; ++c) acc[c] = 0.0f;
#pragma unroll
    for (int dx = 0; dx < 2; ++dx) {
        float wx = dx ? fx : 1.0f - fx;
#pragma unroll
        for (int dy = 0; dy < 2; ++dy) {
            float wy = dy ? fy : 1.0f - fy;
#pragma unroll
            for (int dz = 0; dz < 2; ++dz) {
                float w = wx * wy * (dz ? fz : 1.0f - fz);
                size_t v = (size_t)(((ix + dx) << 12) | ((iy + dy) << 5) | (iz + dz));
                const float* cp = h2 + v * COUT_ + g * 16;
#pragma unroll
                for (int c = 0; c < 16; ++c) acc[c] += w * cp[c];
            }
        }
    }
    float* op = out + (size_t)i * COUT_ + g * 16;
#pragma unroll
    for (int c = 0; c < 16; ++c) op[c] += acc[c];
}

// ==========================================================================
extern "C" void kernel_launch(void* const* d_in, const int* in_sizes, int n_in,
                              void* d_out, int out_size, void* d_ws, size_t ws_size,
                              hipStream_t stream) {
    const float* inputs  = (const float*)d_in[0];
    const float* ptc     = (const float*)d_in[1];
    const int*   vidx    = (const int*)d_in[2];
    const float* ppv     = (const float*)d_in[3];
    const float* mlp_w   = (const float*)d_in[4];
    const float* mlp_b   = (const float*)d_in[5];
    const float* gam     = (const float*)d_in[6];
    const float* bet     = (const float*)d_in[7];
    const float* mu      = (const float*)d_in[8];
    const float* var     = (const float*)d_in[9];
    const float* k1      = (const float*)d_in[10];
    const float* k2      = (const float*)d_in[11];
    float* out = (float*)d_out;

    char* ws = (char*)d_ws;
    float*          gridF = (float*)(ws + 0);                 // 64 MB
    unsigned*       gridB = (unsigned*)(ws + 67108864);       // 32 MB bf16
    unsigned*       maskb = (unsigned*)(ws + 100663296);      //  2 MB
    unsigned short* h1    = (unsigned short*)(ws + 102760448);// 64 MB bf16
    float*          h2    = (float*)(ws + 169869312);         // 128 MB
    unsigned short* wt1   = (unsigned short*)(ws + 304087040);// 108 KB
    unsigned short* wt2   = (unsigned short*)(ws + 304197632);// 216 KB
    unsigned short* wtm   = (unsigned short*)(ws + 304418816);//   4 KB

    (void)hipFuncSetAttribute((const void*)&conv_wmma<32, false>,
                              hipFuncAttributeMaxDynamicSharedMemorySize, 110592);
    (void)hipFuncSetAttribute((const void*)&conv_wmma<64, true>,
                              hipFuncAttributeMaxDynamicSharedMemorySize, 221184);

    (void)hipMemsetAsync(gridF, 0, (size_t)NVOX * CIN_ * sizeof(float), stream);

    prep_weights<<<(27 * 64 * 32 + 27 * 64 * 64 + 64 * 32 + 255) / 256, 256, 0, stream>>>(
        k1, k2, mlp_w, wt1, wt2, wtm);
    scatter_kernel<<<(NPTS * CIN_ + 255) / 256, 256, 0, stream>>>(inputs, vidx, gridF);
    prep_voxels<<<NVOX / 256, 256, 0, stream>>>(gridF, ppv, gridB, maskb);

    mlp_wmma<<<400, 256, 0, stream>>>(inputs, wtm, mlp_b, gam, bet, mu, var, out);

    conv_wmma<32, false><<<1024, 256, 110592, stream>>>(gridB, maskb, wt1, (void*)h1);
    conv_wmma<64, true ><<<1024, 256, 221184, stream>>>((const unsigned*)h1, maskb, wt2, (void*)h2);

    trilinear_add<<<(NPTS * 4 + 255) / 256, 256, 0, stream>>>(h2, ptc, out);
}
